// SparseConv2d_6554120093748
// MI455X (gfx1250) — matrix operand order repaired and checked by reading, not compiled, not run on hardware
//
#include <hip/hip_runtime.h>

#define OC 256
#define IC 256
#define KS 3
#define PADDING 1
#define NB 32
#define H 56
#define W 56
#define KDIM (IC * KS * KS)      // 2304
#define NNZ_CNT 59000
#define NPIX (NB * H * W)        // 100352
#define NWELEM (OC * KDIM)       // 589824
#define NXELEM (NB * IC * H * W) // 25690112
#define HWSZ (H * W)             // 3136

#define BM 128
#define BN 128
#define KC 32
#define NCHUNK (KDIM / KC)       // 72
#define LDA (KC + 8)             // 40 halves -> 80B stride: 20-bank stride, 16B-aligned slices
#define LDB (KC + 8)

typedef __attribute__((ext_vector_type(16))) __bf16 v16bf;
typedef __attribute__((ext_vector_type(8)))  __bf16 v8bf;
typedef __attribute__((ext_vector_type(8)))  float  v8f;
typedef __attribute__((ext_vector_type(4)))  int    v4i;

#define GLOBAL_AS __attribute__((address_space(1)))
#define LDS_AS    __attribute__((address_space(3)))

#if __has_builtin(__builtin_amdgcn_global_load_async_to_lds_b128) && \
    __has_builtin(__builtin_amdgcn_s_wait_asynccnt)
#define USE_ASYNC_LDS 1
#endif

// ---- preprocessing ---------------------------------------------------------

__global__ void zero_f32_kernel(float* __restrict__ p, int n) {
    int i = blockIdx.x * 256 + threadIdx.x;
    if (i < n) p[i] = 0.0f;
}

__global__ void scatter_coo_kernel(const float* __restrict__ val,
                                   const int* __restrict__ idx,
                                   float* __restrict__ w, int nnz) {
    int i = blockIdx.x * 256 + threadIdx.x;
    if (i < nnz) atomicAdd(&w[idx[i]], val[i]);
}

// Weight: f32 [oc][ic*9 + r]  ->  bf16 [oc][r*IC + ic]  ((kh,kw)-major K)
__global__ void permute_w_bf16_kernel(const float* __restrict__ in,
                                      __bf16* __restrict__ out) {
    int i = blockIdx.x * 256 + threadIdx.x;
    if (i < NWELEM) {
        int oc = i / KDIM;
        int t  = i - oc * KDIM;
        int ic = t / 9;
        int r  = t - ic * 9;
        out[oc * KDIM + r * IC + ic] = (__bf16)in[i];
    }
}

// x: f32 NCHW -> bf16 NHWC, 32x32 LDS-tiled transpose (both sides coalesced).
__global__ void nhwc_bf16_kernel(const float* __restrict__ in,
                                 __bf16* __restrict__ out) {
    __shared__ float tile[32][33];
    const int n   = blockIdx.z;
    const int ic0 = blockIdx.y * 32;
    const int hw0 = blockIdx.x * 32;
    const int tx  = threadIdx.x;   // 0..31
    const int ty  = threadIdx.y;   // 0..7
    const float* src = in + (size_t)n * IC * HWSZ;
    #pragma unroll
    for (int j = ty; j < 32; j += 8)
        tile[j][tx] = src[(size_t)(ic0 + j) * HWSZ + hw0 + tx];
    __syncthreads();
    __bf16* dst = out + (size_t)n * HWSZ * IC;
    #pragma unroll
    for (int j = ty; j < 32; j += 8)
        dst[(size_t)(hw0 + j) * IC + ic0 + tx] = (__bf16)tile[tx][j];
}

// ---- fragment loaders (two 16B LDS loads each) -----------------------------

static __device__ __forceinline__ v16bf load_a_frag(const __bf16* row, int kbase) {
    v8bf lo = *(const v8bf*)(row + kbase);        // K = kbase..+7
    v8bf hi = *(const v8bf*)(row + 16 + kbase);   // K = 16+kbase..+7
    return __builtin_shufflevector(lo, hi, 0,1,2,3,4,5,6,7,8,9,10,11,12,13,14,15);
}

static __device__ __forceinline__ v16bf load_b_frag(const __bf16* col, int kbase) {
    v8bf lo = *(const v8bf*)(col + kbase);
    v8bf hi = *(const v8bf*)(col + kbase + 16);
    return __builtin_shufflevector(lo, hi, 0,1,2,3,4,5,6,7,8,9,10,11,12,13,14,15);
}

// ---- implicit-GEMM conv, bf16 WMMA, double-buffered async LDS staging ------
// Block 128 OC x 128 pixels, 8 waves (4 M-bands x 2 N-bands); each wave owns
// a 32x64 sub-tile = 2 A-frags x 4 B-frags -> 8 WMMAs per K-chunk.
// K order: k = r*IC + ic (r = kh*3+kw) -> A address linear in chunk id.

__launch_bounds__(256, 2)
__global__ void sparse_conv_wmma_kernel(const __bf16* __restrict__ xb,   // NHWC bf16
                                        const __bf16* __restrict__ wb,   // [oc][r*IC+ic]
                                        const float* __restrict__ bias,
                                        const __bf16* __restrict__ zpad, // 64B zeros
                                        float* __restrict__ out) {
    __shared__ __bf16 As[2][BM * LDA];   // 2 x 10240 B
    __shared__ __bf16 Bs[2][BN * LDB];   // 2 x 10240 B   ([col][k])

    const int tid  = threadIdx.x;
    const int wave = tid >> 5;
    const int lane = tid & 31;
    const int mw   = wave >> 1;       // 0..3 (32-row band)
    const int nw   = wave & 1;        // 0..1 (64-col band)
    const int m0   = blockIdx.y * BM;
    const int p0   = blockIdx.x * BN; // NPIX % 128 == 0

    // B loader: one pixel column per thread pair, 16-ic (32B) slice each.
    const int bcol = tid >> 1;                // 0..127
    const int bk0  = (tid & 1) * 16;          // 0 or 16
    const int pg   = p0 + bcol;
    const int bn_  = pg / HWSZ;
    const int brem = pg - bn_ * HWSZ;
    const int boh  = brem / W;
    const int bow  = brem - boh * W;
    const int boff = bcol * LDB + bk0;

    // A loader: one 16-half (32B) slice; global address linear in chunk id.
    const int aoff = (tid >> 1) * LDA + (tid & 1) * 16;
    const __bf16* gAbase = wb + (size_t)(m0 + (tid >> 1)) * KDIM + (tid & 1) * 16;

    auto stage = [&](int c, int buf) {
        const __bf16* gA = gAbase + (size_t)c * KC;
        const int r   = c >> 3;
        const int kh  = (r * 11) >> 5;        // r/3 for r in 0..8
        const int kw  = r - kh * 3;
        const int ih  = boh + kh - PADDING;
        const int iw  = bow + kw - PADDING;
        const bool inb = ((unsigned)ih < (unsigned)H) & ((unsigned)iw < (unsigned)W);
        const int icc = (c & 7) * KC;
        const __bf16* gB = inb
            ? xb + (size_t)(pg + (kh - PADDING) * W + (kw - PADDING)) * IC + icc + bk0
            : zpad;
#if defined(USE_ASYNC_LDS)
        __builtin_amdgcn_global_load_async_to_lds_b128(
            (GLOBAL_AS v4i*)gA, (LDS_AS v4i*)&As[buf][aoff], 0, 0);
        __builtin_amdgcn_global_load_async_to_lds_b128(
            (GLOBAL_AS v4i*)gA, (LDS_AS v4i*)&As[buf][aoff], 16, 0);
        // 32B B slice as two b128s; imm offset applies to both global & LDS addr
        __builtin_amdgcn_global_load_async_to_lds_b128(
            (GLOBAL_AS v4i*)gB, (LDS_AS v4i*)&Bs[buf][boff], 0, 0);
        __builtin_amdgcn_global_load_async_to_lds_b128(
            (GLOBAL_AS v4i*)gB, (LDS_AS v4i*)&Bs[buf][boff], 16, 0);
#else
        *(uint4*)&As[buf][aoff] = *(const uint4*)gA;
        *(uint4*)&As[buf][aoff + 8] = *(const uint4*)(gA + 8);
        *(uint4*)&Bs[buf][boff]     = *(const uint4*)gB;
        *(uint4*)&Bs[buf][boff + 8] = *(const uint4*)(gB + 8);
#endif
    };

    // fragment geometry
    const int lmask = lane & 15;
    const int akb   = (lane < 16) ? 0 : 8;
    const int bkb   = (lane < 16) ? 0 : 8;
    const int ar0   = (mw * 32 + lmask) * LDA;
    const int ar1   = ar0 + 16 * LDA;
    int bc[4];
    #pragma unroll
    for (int q = 0; q < 4; ++q)
        bc[q] = (nw * 64 + q * 16 + lmask) * LDB;

    v8f acc[4][2];
    #pragma unroll
    for (int q = 0; q < 4; ++q) { acc[q][0] = (v8f){}; acc[q][1] = (v8f){}; }

    stage(0, 0);
    for (int c = 0; c < NCHUNK; ++c) {
#if defined(USE_ASYNC_LDS)
        __builtin_amdgcn_s_wait_asynccnt(0);   // my chunk-c async loads landed
#endif
        __syncthreads();                        // all waves' chunk-c data visible
        if (c + 1 < NCHUNK) stage(c + 1, (c + 1) & 1);

        const __bf16* Ab = As[c & 1];
        const __bf16* Bb = Bs[c & 1];
        v16bf a0 = load_a_frag(Ab + ar0, akb);
        v16bf a1 = load_a_frag(Ab + ar1, akb);
        #pragma unroll
        for (int q = 0; q < 4; ++q) {
            v16bf b = load_b_frag(Bb + bc[q], bkb);
            acc[q][0] = __builtin_amdgcn_wmma_f32_16x16x32_bf16(false, a0, false, b, (short)0, acc[q][0], false, false);
            acc[q][1] = __builtin_amdgcn_wmma_f32_16x16x32_bf16(false, a1, false, b, (short)0, acc[q][1], false, false);
        }
    }

    // epilogue: C/D layout: VGPR r -> M = r + (lane>=16 ? 8 : 0), N = lane&15
    const int mhi = (lane < 16) ? 0 : 8;
    float bA[8], bB[8];
    #pragma unroll
    for (int r = 0; r < 8; ++r) {
        bA[r] = bias[m0 + mw * 32 + r + mhi];
        bB[r] = bias[m0 + mw * 32 + r + mhi + 16];
    }
    #pragma unroll
    for (int q = 0; q < 4; ++q) {
        const int col = p0 + nw * 64 + q * 16 + lmask;
        const int on  = col / HWSZ;
        const int orr = col - on * HWSZ;
        const int oh  = orr / W;
        const int ow  = orr - oh * W;
        #pragma unroll
        for (int r = 0; r < 8; ++r) {
            const int ocA = m0 + mw * 32 + r + mhi;
            out[(((size_t)on * OC + ocA) * H + oh) * W + ow]      = acc[q][0][r] + bA[r];
            out[(((size_t)on * OC + ocA + 16) * H + oh) * W + ow] = acc[q][1][r] + bB[r];
        }
    }
}

// ---- launch ----------------------------------------------------------------

extern "C" void kernel_launch(void* const* d_in, const int* in_sizes, int n_in,
                              void* d_out, int out_size, void* d_ws, size_t ws_size,
                              hipStream_t stream) {
    const float* x     = (const float*)d_in[0];
    const float* W_val = (const float*)d_in[1];
    const int*   W_idx = (const int*)d_in[2];
    const float* bias  = (const float*)d_in[3];
    float*       out   = (float*)d_out;

    char* ws = (char*)d_ws;
    float*  dense_w = (float*)ws;                                   // 589824 f32
    __bf16* wbf = (__bf16*)(ws + (size_t)NWELEM * 4);               // permuted bf16
    __bf16* xbf = (__bf16*)(ws + (size_t)NWELEM * 4 + (size_t)NWELEM * 2); // NHWC bf16
    __bf16* zpad = (__bf16*)(ws + (size_t)NWELEM * 4 + (size_t)NWELEM * 2
                                + (size_t)NXELEM * 2);              // 64B zero page

    zero_f32_kernel<<<(NWELEM + 255) / 256, 256, 0, stream>>>(dense_w, NWELEM);
    zero_f32_kernel<<<1, 256, 0, stream>>>((float*)zpad, 16);
    scatter_coo_kernel<<<(NNZ_CNT + 255) / 256, 256, 0, stream>>>(W_val, W_idx, dense_w, NNZ_CNT);
    permute_w_bf16_kernel<<<(NWELEM + 255) / 256, 256, 0, stream>>>(dense_w, wbf);

    dim3 tgrid(HWSZ / 32, IC / 32, NB);     // 98 x 8 x 32
    nhwc_bf16_kernel<<<tgrid, dim3(32, 8), 0, stream>>>(x, xbf);

    dim3 grid(NPIX / BN, OC / BM);          // 784 x 2
    sparse_conv_wmma_kernel<<<grid, 256, 0, stream>>>(xbf, wbf, bias, zpad, out);
}
